// SimpleNet_9096740733261
// MI455X (gfx1250) — matrix operand order repaired and checked by reading, not compiled, not run on hardware
//
#include <hip/hip_runtime.h>
#include <hip/hip_bf16.h>
#include <math.h>

#define N_NODES   100000
#define N_EDGES   3200000
#define N_FEAT    128
#define HID       64
#define N_GRAPHS  64
#define ALPHA     0.2f

typedef __attribute__((ext_vector_type(16))) _Float16 v16h;
typedef __attribute__((ext_vector_type(8)))  float    v8f;
typedef __attribute__((ext_vector_type(4)))  float    v4f;

// ---------------------------------------------------------------------------
// GEMM: C[N_NODES x 64] = A[N_NODES x 128] @ W[128 x 64]   (no bias, no act)
// 8 waves / block, each wave computes one 16-row x 64-col tile via
// v_wmma_f32_16x16x32_f16 (4 K-chunks x 4 N-tiles).
// ---------------------------------------------------------------------------
__global__ __launch_bounds__(256) void gemm_k128(const float* __restrict__ A,
                                                 const float* __restrict__ W,
                                                 float* __restrict__ C) {
    constexpr int K    = 128;
    constexpr int LDSW = K + 8;               // halves; 136*2=272B per col, 16B-multiple
    __shared__ _Float16 Wt[HID * LDSW];       // W transposed: [col][k]

    const int tid = threadIdx.x;
    for (int idx = tid; idx < K * HID; idx += 256) {
        const int n = idx & (HID - 1);
        const int k = idx >> 6;
        Wt[n * LDSW + k] = (_Float16)W[k * HID + n];
    }
    __syncthreads();

    const int wave = tid >> 5;
    const int lane = tid & 31;
    const int m    = lane & 15;               // A row within tile / C column within n-tile
    const int kh   = lane >> 4;               // K-half selector

    const long row0 = (long)(blockIdx.x * 8 + wave) * 16;

    v8f acc[4] = {};
    #pragma unroll
    for (int kc = 0; kc < K / 32; ++kc) {
        long arow = row0 + m;
        if (arow >= N_NODES) arow = N_NODES - 1;      // clamp tail, keep EXEC all-1s
        const float* ap = A + arow * K + kc * 32 + kh * 16;
        const v4f a0 = *(const v4f*)(ap + 0);
        const v4f a1 = *(const v4f*)(ap + 4);
        const v4f a2 = *(const v4f*)(ap + 8);
        const v4f a3 = *(const v4f*)(ap + 12);
        v16h afrag;
        #pragma unroll
        for (int i = 0; i < 4; ++i) {
            afrag[i]      = (_Float16)a0[i];
            afrag[4 + i]  = (_Float16)a1[i];
            afrag[8 + i]  = (_Float16)a2[i];
            afrag[12 + i] = (_Float16)a3[i];
        }
        #pragma unroll
        for (int nt = 0; nt < 4; ++nt) {
            const _Float16* bp = &Wt[(nt * 16 + m) * LDSW + kc * 32 + kh * 16];
            v16h bfrag;
            #pragma unroll
            for (int i = 0; i < 16; ++i) bfrag[i] = bp[i];
            acc[nt] = __builtin_amdgcn_wmma_f32_16x16x32_f16(
                false, afrag, false, bfrag, (short)0, acc[nt], false, false);
        }
    }

    #pragma unroll
    for (int nt = 0; nt < 4; ++nt)
        #pragma unroll
        for (int r = 0; r < 8; ++r) {
            const long row = row0 + r + 8 * kh;
            if (row < N_NODES) C[row * HID + nt * 16 + m] = acc[nt][r];
        }
}

// ---------------------------------------------------------------------------
// GEMM: C = act(A[N_NODES x 64] @ W[64 x 64] + bias), act = ReLU if relu!=0
// ---------------------------------------------------------------------------
__global__ __launch_bounds__(256) void gemm_k64(const float* __restrict__ A,
                                                const float* __restrict__ W,
                                                const float* __restrict__ bias,
                                                float* __restrict__ C, int relu) {
    constexpr int K    = 64;
    constexpr int LDSW = K + 8;               // 72 halves = 144B per col, 16B-multiple
    __shared__ _Float16 Wt[HID * LDSW];

    const int tid = threadIdx.x;
    for (int idx = tid; idx < K * HID; idx += 256) {
        const int n = idx & (HID - 1);
        const int k = idx >> 6;
        Wt[n * LDSW + k] = (_Float16)W[k * HID + n];
    }
    __syncthreads();

    const int wave = tid >> 5;
    const int lane = tid & 31;
    const int m    = lane & 15;
    const int kh   = lane >> 4;

    const long row0 = (long)(blockIdx.x * 8 + wave) * 16;

    v8f acc[4] = {};
    #pragma unroll
    for (int kc = 0; kc < K / 32; ++kc) {
        long arow = row0 + m;
        if (arow >= N_NODES) arow = N_NODES - 1;
        const float* ap = A + arow * K + kc * 32 + kh * 16;
        const v4f a0 = *(const v4f*)(ap + 0);
        const v4f a1 = *(const v4f*)(ap + 4);
        const v4f a2 = *(const v4f*)(ap + 8);
        const v4f a3 = *(const v4f*)(ap + 12);
        v16h afrag;
        #pragma unroll
        for (int i = 0; i < 4; ++i) {
            afrag[i]      = (_Float16)a0[i];
            afrag[4 + i]  = (_Float16)a1[i];
            afrag[8 + i]  = (_Float16)a2[i];
            afrag[12 + i] = (_Float16)a3[i];
        }
        #pragma unroll
        for (int nt = 0; nt < 4; ++nt) {
            const _Float16* bp = &Wt[(nt * 16 + m) * LDSW + kc * 32 + kh * 16];
            v16h bfrag;
            #pragma unroll
            for (int i = 0; i < 16; ++i) bfrag[i] = bp[i];
            acc[nt] = __builtin_amdgcn_wmma_f32_16x16x32_f16(
                false, afrag, false, bfrag, (short)0, acc[nt], false, false);
        }
    }

    #pragma unroll
    for (int nt = 0; nt < 4; ++nt) {
        const int col = nt * 16 + m;
        const float b = bias[col];
        #pragma unroll
        for (int r = 0; r < 8; ++r) {
            const long row = row0 + r + 8 * kh;
            if (row < N_NODES) {
                float v = acc[nt][r] + b;
                if (relu) v = fmaxf(v, 0.0f);
                C[row * HID + col] = v;
            }
        }
    }
}

// --------------------------- elementwise / graph kernels -------------------
__global__ void init_deg(float* deg) {
    const int i = blockIdx.x * blockDim.x + threadIdx.x;
    if (i < N_NODES) deg[i] = 1.0f;             // self-loop
}

__global__ void deg_edges(const int* __restrict__ ei, float* deg) {
    const long e = (long)blockIdx.x * blockDim.x + threadIdx.x;
    if (e < N_EDGES) atomicAdd(&deg[ei[N_EDGES + e]], 1.0f);
}

__global__ void deg_to_dinv(float* deg) {
    const int i = blockIdx.x * blockDim.x + threadIdx.x;
    if (i < N_NODES) deg[i] = rsqrtf(deg[i]);
}

// agg = dinv^2 * h   (self-loop message)
__global__ void self_init(const float* __restrict__ dinv,
                          const float* __restrict__ h, float* __restrict__ agg) {
    const long gid = (long)blockIdx.x * blockDim.x + threadIdx.x;
    if (gid < (long)N_NODES * HID) {
        const long n = gid >> 6;
        const float di = dinv[n];
        agg[gid] = di * di * h[gid];
    }
}

// agg[dst] += dinv[src]*dinv[dst] * h[src]; one thread per (edge, feature)
__global__ void edge_agg(const int* __restrict__ ei, const float* __restrict__ dinv,
                         const float* __restrict__ h, float* __restrict__ agg) {
    const long gid = (long)blockIdx.x * blockDim.x + threadIdx.x;
    if (gid >= (long)N_EDGES * HID) return;
    const int  f = (int)(gid & (HID - 1));
    const long e = gid >> 6;
    const int  s = ei[e];
    const int  d = ei[N_EDGES + e];
    const float norm = dinv[s] * dinv[d];
    atomicAdd(&agg[(long)d * HID + f], norm * h[(long)s * HID + f]);
}

// h = leaky_relu(agg + conv_b)
__global__ void bias_leaky(const float* __restrict__ agg,
                           const float* __restrict__ b, float* __restrict__ h) {
    const long gid = (long)blockIdx.x * blockDim.x + threadIdx.x;
    if (gid < (long)N_NODES * HID) {
        const float v = agg[gid] + b[gid & (HID - 1)];
        h[gid] = v >= 0.0f ? v : ALPHA * v;
    }
}

__global__ void zero_pool(float* pooled, float* cnts) {
    const int i = threadIdx.x;
    if (i < N_GRAPHS) { pooled[i] = 0.0f; cnts[i] = 0.0f; }
}

// score[n] = h[n]·w3 + b3; segment-sum into pooled/cnts
__global__ void score_pool(const float* __restrict__ h, const float* __restrict__ w3,
                           const float* __restrict__ b3, const int* __restrict__ batch,
                           float* pooled, float* cnts) {
    const int n = blockIdx.x * blockDim.x + threadIdx.x;
    if (n >= N_NODES) return;
    float s = b3[0];
    const float* r = h + (long)n * HID;
    #pragma unroll
    for (int i = 0; i < HID; ++i) s = fmaf(r[i], w3[i], s);
    const int g = batch[n];
    atomicAdd(&pooled[g], s);
    atomicAdd(&cnts[g], 1.0f);
}

__global__ void finalize(const float* pooled, const float* cnts, float* out) {
    const int g = threadIdx.x;
    if (g < N_GRAPHS) {
        const float mean = pooled[g] / cnts[g];
        out[g] = 1.0f / (1.0f + expf(-mean));
    }
}

// ---------------------------------------------------------------------------
extern "C" void kernel_launch(void* const* d_in, const int* in_sizes, int n_in,
                              void* d_out, int out_size, void* d_ws, size_t ws_size,
                              hipStream_t stream) {
    const float* x       = (const float*)d_in[0];
    const int*   ei      = (const int*)  d_in[1];
    const int*   batch   = (const int*)  d_in[2];
    const float* conv_w  = (const float*)d_in[3];
    const float* conv_b  = (const float*)d_in[4];
    const float* w1      = (const float*)d_in[5];
    const float* b1      = (const float*)d_in[6];
    const float* w2      = (const float*)d_in[7];
    const float* b2      = (const float*)d_in[8];
    const float* w3      = (const float*)d_in[9];
    const float* b3      = (const float*)d_in[10];
    float*       out     = (float*)d_out;

    // workspace layout (all float32)
    float* h      = (float*)d_ws;                       // [N, 64]
    float* agg    = h    + (size_t)N_NODES * HID;       // [N, 64]
    float* dinv   = agg  + (size_t)N_NODES * HID;       // [N]
    float* pooled = dinv + N_NODES;                     // [G]
    float* cnts   = pooled + N_GRAPHS;                  // [G]

    const int gemm_blocks = (N_NODES + 127) / 128;      // 8 waves * 16 rows per block
    const long nh   = (long)N_NODES * HID;
    const long neh  = (long)N_EDGES * HID;

    // 1. h = X @ conv_w  (WMMA)
    gemm_k128<<<gemm_blocks, 256, 0, stream>>>(x, conv_w, h);

    // 2. degree -> dinv
    init_deg  <<<(N_NODES + 255) / 256, 256, 0, stream>>>(dinv);
    deg_edges <<<(N_EDGES + 255) / 256, 256, 0, stream>>>(ei, dinv);
    deg_to_dinv<<<(N_NODES + 255) / 256, 256, 0, stream>>>(dinv);

    // 3. symmetric-normalized aggregation (L2-resident atomics)
    self_init<<<(int)((nh + 255) / 256), 256, 0, stream>>>(dinv, h, agg);
    edge_agg <<<(int)((neh + 255) / 256), 256, 0, stream>>>(ei, dinv, h, agg);

    // 4. h = leaky_relu(agg + conv_b)
    bias_leaky<<<(int)((nh + 255) / 256), 256, 0, stream>>>(agg, conv_b, h);

    // 5. MLP layers (WMMA, fused bias+ReLU)
    gemm_k64<<<gemm_blocks, 256, 0, stream>>>(h,   w1, b1, agg, 1);
    gemm_k64<<<gemm_blocks, 256, 0, stream>>>(agg, w2, b2, h,   1);

    // 6. final projection + segment-mean pool + sigmoid
    zero_pool <<<1, 128, 0, stream>>>(pooled, cnts);
    score_pool<<<(N_NODES + 255) / 256, 256, 0, stream>>>(h, w3, b3, batch, pooled, cnts);
    finalize  <<<1, 64, 0, stream>>>(pooled, cnts, out);
}
